// MultiHeadAttention_4260607558031
// MI455X (gfx1250) — matrix operand order repaired and checked by reading, not compiled, not run on hardware
//
#include <hip/hip_runtime.h>

typedef _Float16 f16;
typedef __attribute__((ext_vector_type(16))) _Float16 v16h;
typedef __attribute__((ext_vector_type(8)))  _Float16 v8h;
typedef __attribute__((ext_vector_type(8)))  float    v8f;

#define DEVI static __device__ __forceinline__

// ---- problem constants (fixed by reference setup) ----
constexpr int BB = 2, SS = 2048, EE = 512, HH = 8, HD = 64, E3 = 1536;
constexpr int MM = BB * SS;            // 4096 rows of x / qkv / vals

// ---- workspace layout (bytes) ----
constexpr size_t OFF_XH   = 0;                                  // 4096x512 f16 (row major)
constexpr size_t OFF_WQ   = OFF_XH  + (size_t)MM * EE * 2;      // qkv_w f16, N(1536) x K(512) row major
constexpr size_t OFF_WO   = OFF_WQ  + (size_t)E3 * EE * 2;      // o_w   f16, N(512)  x K(512) row major
constexpr size_t OFF_Q    = OFF_WO  + (size_t)EE * EE * 2;      // [b][h][s][64] f16
constexpr size_t OFF_K    = OFF_Q   + (size_t)BB*HH*SS*HD*2;    // [b][h][s][64] f16 (d contiguous)
constexpr size_t OFF_VT   = OFF_K   + (size_t)BB*HH*SS*HD*2;    // [b][h][64][s] f16 (s contiguous)
constexpr size_t OFF_VALS = OFF_VT  + (size_t)BB*HH*SS*HD*2;    // 4096x512 f16 (row major)

// ---- fragment loads: all contiguous 16-byte chunks ----
DEVI v8h ld8(const f16* p) { return *(const v8h*)p; }
DEVI v16h cat16(v8h lo, v8h hs) {
  v16h r;
#pragma unroll
  for (int i = 0; i < 8; ++i) { r[i] = lo[i]; r[8 + i] = hs[i]; }
  return r;
}
// A 16x32: lane = row (lane&15); halves K = hi*16 + 0..15 (one contiguous run)
DEVI v16h afrag(const f16* p /* row ptr + k base */, int hi) {
  return cat16(ld8(p + hi * 16), ld8(p + hi * 16 + 8));
}
// B 32x16: lane = col (lane&15); halves K = hi*16 + 0..15 (one contiguous run)
DEVI v16h bfragc(const f16* p, int hi) {
  const f16* q = p + hi * 16;
  return cat16(ld8(q), ld8(q + 8));
}
DEVI v8f wmma16(v16h a, v16h b, v8f c) {
  return __builtin_amdgcn_wmma_f32_16x16x32_f16(false, a, false, b, (short)0, c,
                                                false, false);
}

// ---- fp32 -> f16 elementwise ----
__global__ void k_cvt(const float* __restrict__ s, f16* __restrict__ d, int n) {
  int i = blockIdx.x * blockDim.x + threadIdx.x;
  int st = gridDim.x * blockDim.x;
  for (; i < n; i += st) d[i] = (f16)s[i];
}

// ---- software-pipelined 32x64-per-wave GEMM core: C(32x64) += A(32xK) @ B(KxN)^T ----
// ar0/ar1: lane's A rows (m and m+16), br: lane's B row (N x K row-major), acc[8].
DEVI void gemm_core(const f16* __restrict__ ar0, const f16* __restrict__ ar1,
                    const f16* __restrict__ br, int hi, v8f acc[8]) {
  v16h a[2][2], b[2][4];
  a[0][0] = afrag(ar0, hi);
  a[0][1] = afrag(ar1, hi);
#pragma unroll
  for (int t = 0; t < 4; ++t) b[0][t] = bfragc(br + (size_t)(t * 16) * EE, hi);
#pragma unroll
  for (int i = 0; i < 16; ++i) {
    const int cur = i & 1, nxt = cur ^ 1;
    const int kn = i * 32 + 32;
    if (i < 15) {                       // prefetch next K-step while computing
      a[nxt][0] = afrag(ar0 + kn, hi);
      a[nxt][1] = afrag(ar1 + kn, hi);
#pragma unroll
      for (int t = 0; t < 4; ++t)
        b[nxt][t] = bfragc(br + (size_t)(t * 16) * EE + kn, hi);
    }
#pragma unroll
    for (int t = 0; t < 4; ++t) acc[t]     = wmma16(a[cur][0], b[cur][t], acc[t]);
#pragma unroll
    for (int t = 0; t < 4; ++t) acc[4 + t] = wmma16(a[cur][1], b[cur][t], acc[4 + t]);
  }
}

// ---- QKV projection: qkv = x @ qkv_w^T + b ; scatter to Q / K / V^T (f16) ----
__global__ void k_qkv_gemm(const f16* __restrict__ xh, const f16* __restrict__ wh,
                           const float* __restrict__ bias, f16* __restrict__ Q,
                           f16* __restrict__ Kb, f16* __restrict__ Vt) {
  int wid = threadIdx.x >> 5;
  int lane = threadIdx.x & 31, ln = lane & 15, hi = lane >> 4;
  int wg = blockIdx.x * 4 + wid;         // 0 .. 3071
  int mt = wg / 24, nt = wg - mt * 24;
  int mb = mt * 32, nb = nt * 64;
  v8f z = {};
  v8f acc[8] = {z, z, z, z, z, z, z, z};
  const f16* ar0 = xh + (size_t)(mb + ln) * EE;
  gemm_core(ar0, ar0 + (size_t)16 * EE, wh + (size_t)(nb + ln) * EE, hi, acc);
#pragma unroll
  for (int u = 0; u < 2; ++u) {
#pragma unroll
    for (int t = 0; t < 4; ++t) {
      int n = nb + t * 16 + ln;
      float bv = bias[n];
      int h = n / 192, rr = n - h * 192;
#pragma unroll
      for (int r = 0; r < 8; ++r) {
        int m = mb + u * 16 + r + hi * 8;
        int bidx = m >> 11, s = m & (SS - 1);
        size_t bh = (size_t)(bidx * HH + h);
        f16 hv = (f16)(acc[u * 4 + t][r] + bv);
        if (rr < 64)       Q[(bh * SS + s) * HD + rr] = hv;
        else if (rr < 128) Kb[(bh * SS + s) * HD + (rr - 64)] = hv;
        else               Vt[(bh * HD + (rr - 128)) * SS + s] = hv;
      }
    }
  }
}

// ---- banded attention with online softmax; one 16-query tile per wave ----
__global__ void k_attn(const f16* __restrict__ Q, const f16* __restrict__ Kmat,
                       const f16* __restrict__ Vt, const int* __restrict__ pm,
                       f16* __restrict__ vals) {
  __shared__ __align__(32) f16 pbuf[4][16 * 32];
  int wid = threadIdx.x >> 5;
  int lane = threadIdx.x & 31, ln = lane & 15, hi = lane >> 4;
  int wg = blockIdx.x * 4 + wid;        // 0 .. 2047
  int bh = wg >> 7;                      // (b*H + h)
  int qt = wg & 127;
  int b = bh >> 3, h = bh & 7;
  int s0g = qt * 16;

  const f16* Qb  = Q    + (size_t)bh * SS * HD;
  const f16* Kbb = Kmat + (size_t)bh * SS * HD;
  const f16* Vtb = Vt   + (size_t)bh * HD * SS;
  const int* pmb = pm + b * SS;

  const f16* qrow = Qb + (size_t)(s0g + ln) * HD;
  v16h a0 = afrag(qrow + 0, hi);
  v16h a1 = afrag(qrow + 32, hi);

  float mrun[8], lrun[8];
  v8f z = {};
  v8f oa[4] = {z, z, z, z};
#pragma unroll
  for (int r = 0; r < 8; ++r) { mrun[r] = -3.0e38f; lrun[r] = 0.f; }

  int jt0 = (s0g >= 128) ? ((s0g - 128) >> 4) : 0;
  int jt1 = (s0g + 143) >> 4; if (jt1 > 127) jt1 = 127;
  f16* pb = pbuf[wid];

  for (int jt = jt0; jt <= jt1; jt += 2) {
    int kb2 = jt * 16;
    // --- hoisted loads: V fragments (used only after softmax) + both K sub-tiles ---
    v16h bv[4];
#pragma unroll
    for (int t = 0; t < 4; ++t)
      bv[t] = bfragc(Vtb + (size_t)(t * 16 + ln) * SS + kb2, hi);

    int tcol0 = kb2 + ln,       tc0 = tcol0 < (SS - 1) ? tcol0 : (SS - 1);
    int tcol1 = kb2 + 16 + ln,  tc1 = tcol1 < (SS - 1) ? tcol1 : (SS - 1);
    const f16* kr0 = Kbb + (size_t)tc0 * HD;
    const f16* kr1 = Kbb + (size_t)tc1 * HD;
    v16h bk00 = bfragc(kr0, hi), bk01 = bfragc(kr0 + 32, hi);
    v16h bk10 = bfragc(kr1, hi), bk11 = bfragc(kr1 + 32, hi);
    bool pmok0 = (pmb[tc0] != 0), pmok1 = (pmb[tc1] != 0);

    // --- score WMMAs ---
    v8f sv0 = {}, sv1 = {};
    sv0 = wmma16(a0, bk00, sv0);
    sv0 = wmma16(a1, bk01, sv0);
    sv1 = wmma16(a0, bk10, sv1);
    sv1 = wmma16(a1, bk11, sv1);

    // --- mask + scale ---
    float sc[2][8];
    unsigned okm[2] = {0u, 0u};
#pragma unroll
    for (int r = 0; r < 8; ++r) {
      int qr = s0g + r + hi * 8;
      bool ok0 = pmok0 && (tcol0 < SS) && (tcol0 >= qr - 128) && (tcol0 <= qr + 128);
      bool ok1 = pmok1 && (tcol1 < SS) && (tcol1 >= qr - 128) && (tcol1 <= qr + 128);
      sc[0][r] = ok0 ? sv0[r] * 0.125f : -3.0e38f;   // 1/sqrt(64)
      sc[1][r] = ok1 ? sv1[r] * 0.125f : -3.0e38f;
      okm[0] |= ok0 ? (1u << r) : 0u;
      okm[1] |= ok1 ? (1u << r) : 0u;
    }
    // --- per-row running max (rows per-VGPR, columns across each 16-lane half) ---
    float mx[8];
#pragma unroll
    for (int r = 0; r < 8; ++r) {
      float v = fmaxf(fmaxf(sc[0][r], sc[1][r]), mrun[r]);
      v = fmaxf(v, __shfl_xor(v, 1, 32));
      v = fmaxf(v, __shfl_xor(v, 2, 32));
      v = fmaxf(v, __shfl_xor(v, 4, 32));
      v = fmaxf(v, __shfl_xor(v, 8, 32));
      mx[r] = v;
    }
#pragma unroll
    for (int r = 0; r < 8; ++r) {
      float alpha = __expf(mrun[r] - mx[r]);
      float p0 = (okm[0] >> r & 1) ? __expf(sc[0][r] - mx[r]) : 0.f;
      float p1 = (okm[1] >> r & 1) ? __expf(sc[1][r] - mx[r]) : 0.f;
      float rs = p0 + p1;
      rs += __shfl_xor(rs, 1, 32);
      rs += __shfl_xor(rs, 2, 32);
      rs += __shfl_xor(rs, 4, 32);
      rs += __shfl_xor(rs, 8, 32);
      lrun[r] = lrun[r] * alpha + rs;
      mrun[r] = mx[r];
#pragma unroll
      for (int t = 0; t < 4; ++t) oa[t][r] = oa[t][r] * alpha;
      pb[(r + hi * 8) * 32 + ln]      = (f16)p0;   // C-layout -> row-major P in LDS
      pb[(r + hi * 8) * 32 + 16 + ln] = (f16)p1;
    }
    // --- P@V with preloaded V fragments ---
    v16h ap = afrag(pb + (size_t)ln * 32, hi);
#pragma unroll
    for (int t = 0; t < 4; ++t) oa[t] = wmma16(ap, bv[t], oa[t]);
  }
  // normalize + store to [b][s][h*64+d] f16
#pragma unroll
  for (int t = 0; t < 4; ++t) {
#pragma unroll
    for (int r = 0; r < 8; ++r) {
      float l = lrun[r];
      float ov = (l > 0.f) ? oa[t][r] / l : 0.f;
      int m = s0g + r + hi * 8;
      vals[(size_t)(b * SS + m) * EE + h * HD + t * 16 + ln] = (f16)ov;
    }
  }
}

// ---- output projection: out = vals @ o_w^T + o_b (fp32 out) ----
__global__ void k_out_gemm(const f16* __restrict__ vals, const f16* __restrict__ woh,
                           const float* __restrict__ ob, float* __restrict__ out) {
  int wid = threadIdx.x >> 5;
  int lane = threadIdx.x & 31, ln = lane & 15, hi = lane >> 4;
  int wg = blockIdx.x * 4 + wid;         // 0 .. 1023
  int mt = wg >> 3, nt = wg & 7;
  int mb = mt * 32, nb = nt * 64;
  v8f z = {};
  v8f acc[8] = {z, z, z, z, z, z, z, z};
  const f16* ar0 = vals + (size_t)(mb + ln) * EE;
  gemm_core(ar0, ar0 + (size_t)16 * EE, woh + (size_t)(nb + ln) * EE, hi, acc);
#pragma unroll
  for (int u = 0; u < 2; ++u) {
#pragma unroll
    for (int t = 0; t < 4; ++t) {
      int n = nb + t * 16 + ln;
      float bv = ob[n];
#pragma unroll
      for (int r = 0; r < 8; ++r) {
        int m = mb + u * 16 + r + hi * 8;
        out[(size_t)m * EE + n] = acc[u * 4 + t][r] + bv;
      }
    }
  }
}

extern "C" void kernel_launch(void* const* d_in, const int* in_sizes, int n_in,
                              void* d_out, int out_size, void* d_ws, size_t ws_size,
                              hipStream_t stream) {
  const float* x     = (const float*)d_in[0];
  const int*   pmask = (const int*)d_in[1];
  const float* qkv_w = (const float*)d_in[2];
  const float* qkv_b = (const float*)d_in[3];
  const float* o_w   = (const float*)d_in[4];
  const float* o_b   = (const float*)d_in[5];
  float* out = (float*)d_out;

  char* ws = (char*)d_ws;
  f16* xh   = (f16*)(ws + OFF_XH);
  f16* wq   = (f16*)(ws + OFF_WQ);
  f16* wo   = (f16*)(ws + OFF_WO);
  f16* Q    = (f16*)(ws + OFF_Q);
  f16* Kb   = (f16*)(ws + OFF_K);
  f16* Vt   = (f16*)(ws + OFF_VT);
  f16* vals = (f16*)(ws + OFF_VALS);

  k_cvt<<<1024, 256, 0, stream>>>(x, xh, MM * EE);
  k_cvt<<<768, 256, 0, stream>>>(qkv_w, wq, E3 * EE);
  k_cvt<<<256, 256, 0, stream>>>(o_w, wo, EE * EE);
  // 128 M32-tiles * 24 N64-tiles = 3072 waves, 4 waves/block
  k_qkv_gemm<<<768, 128, 0, stream>>>(xh, wq, qkv_b, Q, Kb, Vt);
  // 2 * 8 * 128 query tiles = 2048 waves
  k_attn<<<512, 128, 0, stream>>>(Q, Kb, Vt, pmask, vals);
  // 128 M32-tiles * 8 N64-tiles = 1024 waves
  k_out_gemm<<<256, 128, 0, stream>>>(vals, wo, o_b, out);
}
